// DeepARDecoder_13048110645728
// MI455X (gfx1250) — matrix operand order repaired and checked by reading, not compiled, not run on hardware
//
#include <hip/hip_runtime.h>
#include <hip/hip_bf16.h>

// ---------------- problem constants (match reference) ----------------
#define NL   3      // layers
#define NB   512    // batch
#define ND   1024   // d_model H
#define ND2  2048   // packed K extent (x | hprev), (Wih_row | Whh_row)
#define NT   8      // targets
#define NTAU 96     // decode steps

typedef _Float16 v16h __attribute__((ext_vector_type(16)));
typedef _Float16 v8h  __attribute__((ext_vector_type(8)));
typedef float    v8f  __attribute__((ext_vector_type(8)));

// Build a 16-half WMMA fragment from two contiguous 16B chunks.
__device__ __forceinline__ v16h load_frag2(const _Float16* p0, const _Float16* p1) {
    v8h a = *(const v8h*)p0;
    v8h b = *(const v8h*)p1;
    return __builtin_shufflevector(a, b, 0,1,2,3,4,5,6,7,8,9,10,11,12,13,14,15);
}

__device__ __forceinline__ float fast_sigmoid(float x) {
    return 1.0f / (1.0f + __expf(-x));
}
__device__ __forceinline__ float fast_tanh(float x) {
    return 1.0f - 2.0f / (__expf(2.0f * x) + 1.0f);
}

// Async copy 128B (one packed-weight row chunk) global -> LDS.
// IOFFSET is added to BOTH the LDS address (VDST) and the global address.
__device__ __forceinline__ void async_stage_row128(uint32_t lds_addr, const _Float16* gptr) {
    asm volatile("global_load_async_to_lds_b128 %0, %1, off offset:0"
                 :: "v"(lds_addr), "v"(gptr) : "memory");
    asm volatile("global_load_async_to_lds_b128 %0, %1, off offset:16"
                 :: "v"(lds_addr), "v"(gptr) : "memory");
    asm volatile("global_load_async_to_lds_b128 %0, %1, off offset:32"
                 :: "v"(lds_addr), "v"(gptr) : "memory");
    asm volatile("global_load_async_to_lds_b128 %0, %1, off offset:48"
                 :: "v"(lds_addr), "v"(gptr) : "memory");
    asm volatile("global_load_async_to_lds_b128 %0, %1, off offset:64"
                 :: "v"(lds_addr), "v"(gptr) : "memory");
    asm volatile("global_load_async_to_lds_b128 %0, %1, off offset:80"
                 :: "v"(lds_addr), "v"(gptr) : "memory");
    asm volatile("global_load_async_to_lds_b128 %0, %1, off offset:96"
                 :: "v"(lds_addr), "v"(gptr) : "memory");
    asm volatile("global_load_async_to_lds_b128 %0, %1, off offset:112"
                 :: "v"(lds_addr), "v"(gptr) : "memory");
}

__device__ __forceinline__ void wait_async8() {
    asm volatile("s_wait_asynccnt 0x8" ::: "memory");
}
__device__ __forceinline__ void wait_async0() {
    asm volatile("s_wait_asynccnt 0x0" ::: "memory");
}

// ---------------- init / conversion kernels ----------------
// Pack weights: Wp[l][r][0:1024] = Wih[l][r][:], Wp[l][r][1024:2048] = Whh[l][r][:]
__global__ void k_pack_weights(const float* __restrict__ Wih, const float* __restrict__ Whh,
                               _Float16* __restrict__ Wp, int n) {
    for (int i = blockIdx.x * blockDim.x + threadIdx.x; i < n; i += gridDim.x * blockDim.x) {
        const int col = i & (ND2 - 1);
        const int row = i >> 11;                       // global row in [0, NL*4*ND)
        const float v = (col < ND) ? Wih[(size_t)row * ND + col]
                                   : Whh[(size_t)row * ND + (col - ND)];
        Wp[i] = (_Float16)v;
    }
}

__global__ void k_bias_combine(const float* __restrict__ bih, const float* __restrict__ bhh,
                               float* __restrict__ out, int n) {
    int i = blockIdx.x * blockDim.x + threadIdx.x;
    if (i < n) out[i] = bih[i] + bhh[i];
}

// act layout: [NL][NB][ND2] f16 with cols 0..ND-1 = x, ND..2ND-1 = hprev
__global__ void k_init_act(const float* __restrict__ hidden, const float* __restrict__ cell,
                           _Float16* __restrict__ act0, float* __restrict__ c, int n) {
    int i = blockIdx.x * blockDim.x + threadIdx.x;
    if (i >= n) return;
    const int l   = i >> 19;           // / (NB*ND)
    const int m   = (i >> 10) & (NB - 1);
    const int col = i & (ND - 1);
    const float hv = hidden[i];
    act0[((size_t)l * NB + m) * ND2 + ND + col] = (_Float16)hv;   // hprev half
    if (l == NL - 1)
        act0[(size_t)m * ND2 + col] = (_Float16)hv;               // x half of layer 0
    c[i] = cell[i];
}

// ---------------- fused LSTM layer kernel ----------------
// gates = [x|hprev] @ [Wih|Whh]^T + bias over packed K=2048, then LSTM cell.
// Workgroup = 256 threads = 8 waves: 4 (M) x 2 (N). WG tile: M=64, N=64 per gate.
// Weights staged via async-to-LDS in a 3-deep ring (rounds of K=64, 32 rounds),
// prefetch distance 2, one workgroup barrier per round.
// Grid: (NB/64, ND/64).
__global__ __launch_bounds__(256, 2)
void k_lstm_layer(const _Float16* __restrict__ actA,   // [NB, ND2] rows = x|hprev
                  const _Float16* __restrict__ Wp,     // [4*ND, ND2] packed f16
                  const float*    __restrict__ bias,   // [4*ND]
                  float*          __restrict__ c,      // [NB, ND] in/out
                  _Float16*       __restrict__ hdst_x, // consumer x-half base ([NB, ND2])
                  _Float16*       __restrict__ hdst_h) // next-step hprev-half base
{
    __shared__ _Float16 smemB[3 * 16384];   // 3 x 32KB ring

    const int tid  = threadIdx.x;
    const int lane = tid & 31;
    const int wid  = tid >> 5;
    const int hlf  = lane >> 4;
    const int l16  = lane & 15;
    const int m0   = blockIdx.x * 64 + (wid >> 1) * 16;   // M strip for this wave
    const int nwg0 = blockIdx.y * 64;                     // WG column block within a gate
    const int nw   = (wid & 1) * 32;                      // wave column sub-block

    v8f acc[4][2];
#pragma unroll
    for (int g = 0; g < 4; ++g)
#pragma unroll
        for (int s = 0; s < 2; ++s)
            acc[g][s] = (v8f){0.f,0.f,0.f,0.f,0.f,0.f,0.f,0.f};

    // A-fragment row base (packed [x|hprev], stride ND2)
    const _Float16* arow = actA + (size_t)(m0 + l16) * ND2;

    // Staging: thread tid owns packed-weight row (g*ND + nwg0 + c), g=tid>>6, c=tid&63.
    const int g_st = tid >> 6;
    const int c_st = tid & 63;
    const _Float16* gw = Wp + ((size_t)g_st * ND + nwg0 + c_st) * ND2;
    const uint32_t sb = (uint32_t)(uintptr_t)(&smemB[0]);
    const uint32_t lds_slot = (uint32_t)(tid * 128);

    // Fragment read base in LDS: row = g*64 + (nw + s*16 + l16), row stride 64 halves.
    const _Float16* lrow = &smemB[0] + (size_t)(nw + l16) * 64 + hlf * 16;

    auto stage = [&](int j) {
        const uint32_t ld = sb + (uint32_t)((j % 3) * 32768) + lds_slot;
        async_stage_row128(ld, gw + j * 64);
    };
    auto compute = [&](int r) {
        const _Float16* lb = lrow + (r % 3) * 16384;
        const _Float16* ar = arow + r * 64;
#pragma unroll
        for (int u = 0; u < 2; ++u) {
            const _Float16* ap = ar + u * 32;
            v16h a = load_frag2(ap + hlf * 16, ap + hlf * 16 + 8);
#pragma unroll
            for (int g = 0; g < 4; ++g)
#pragma unroll
                for (int s = 0; s < 2; ++s) {
                    const _Float16* lp = lb + (g * 64 + s * 16) * 64 + u * 32;
                    v16h b = load_frag2(lp, lp + 8);
                    acc[g][s] = __builtin_amdgcn_wmma_f32_16x16x32_f16(
                        false, a, false, b, (short)0, acc[g][s], false, false);
                }
        }
    };

    // prologue: prefetch rounds 0 and 1
    stage(0);
    stage(1);
    for (int r = 0; r < 30; ++r) {
        wait_async8();      // in-order per-wave completion: oldest round landed
        __syncthreads();    // all waves' stages landed; previous round's reads done
        compute(r);
        stage(r + 2);
    }
    wait_async8(); __syncthreads(); compute(30);
    wait_async0(); __syncthreads(); compute(31);

    // ---- epilogue: bias + LSTM cell on accumulator fragments ----
    // C/D layout: VGPR r, lanes 0-15 -> M=m0+r, N=lane; lanes 16-31 -> M=m0+r+8, N=lane-16.
#pragma unroll
    for (int s = 0; s < 2; ++s) {
        const int ncol = nwg0 + nw + s * 16 + l16;          // column within [0, ND)
        const float bi = bias[0 * ND + ncol];
        const float bf = bias[1 * ND + ncol];
        const float bg = bias[2 * ND + ncol];
        const float bo = bias[3 * ND + ncol];
#pragma unroll
        for (int r = 0; r < 8; ++r) {
            const int m = m0 + r + hlf * 8;
            const size_t idxc = (size_t)m * ND + ncol;
            const size_t idxa = (size_t)m * ND2 + ncol;
            const float iv = fast_sigmoid(acc[0][s][r] + bi);
            const float fv = fast_sigmoid(acc[1][s][r] + bf);
            const float gv = fast_tanh   (acc[2][s][r] + bg);
            const float ov = fast_sigmoid(acc[3][s][r] + bo);
            const float cn = fv * c[idxc] + iv * gv;
            c[idxc] = cn;
            const _Float16 hn = (_Float16)(ov * fast_tanh(cn));
            hdst_x[idxa] = hn;       // consumer layer's x half
            hdst_h[idxa] = hn;       // own hprev half for next step
        }
    }
}

// ---------------- per-step projection: mu / sigma for one t ----------------
__global__ __launch_bounds__(256)
void k_project(const _Float16* __restrict__ hy,   // [NB, ND2]: layer-0 x-half = y_t
               const float* __restrict__ W1, const float* __restrict__ b1,
               const float* __restrict__ W2, const float* __restrict__ b2,
               float* __restrict__ mu, float* __restrict__ sigma, int t)
{
    const int tid = blockIdx.x * blockDim.x + threadIdx.x;   // 0 .. NB*NT-1
    const int b  = tid >> 3;
    const int tg = tid & 7;
    const _Float16* hr = hy + (size_t)b * ND2;
    const float* w1 = W1 + (size_t)tg * ND;
    const float* w2 = W2 + (size_t)tg * ND;
    float a1 = 0.f, a2 = 0.f;
    for (int k = 0; k < ND; k += 8) {
        v8h hv = *(const v8h*)(hr + k);
#pragma unroll
        for (int j = 0; j < 8; ++j) {
            const float h = (float)hv[j];
            a1 += h * w1[k + j];
            a2 += h * w2[k + j];
        }
    }
    const size_t o = ((size_t)b * NTAU + t) * NT + tg;
    mu[o] = a1 + b1[tg];
    const float z  = 2.0f * (a2 + b2[tg]);                 // BETA * pre-act
    const float sp = (z > 20.0f) ? z : log1pf(__expf(z));  // softplus
    sigma[o] = 0.5f * sp;                                  // / BETA
}

// ---------------- launcher ----------------
extern "C" void kernel_launch(void* const* d_in, const int* in_sizes, int n_in,
                              void* d_out, int out_size, void* d_ws, size_t ws_size,
                              hipStream_t stream) {
    const float* hidden = (const float*)d_in[0];   // [L,B,H]
    const float* cell   = (const float*)d_in[1];   // [L,B,H]
    const float* W_ih   = (const float*)d_in[2];   // [L,4H,H]
    const float* W_hh   = (const float*)d_in[3];   // [L,4H,H]
    const float* b_ih   = (const float*)d_in[4];   // [L,4H]
    const float* b_hh   = (const float*)d_in[5];   // [L,4H]
    const float* W1     = (const float*)d_in[6];   // [T,H]
    const float* b1     = (const float*)d_in[7];   // [T]
    const float* W2     = (const float*)d_in[8];   // [T,H]
    const float* b2     = (const float*)d_in[9];   // [T]

    float* mu    = (float*)d_out;                         // [B,TAU,T]
    float* sigma = mu + (size_t)NB * NTAU * NT;           // [B,TAU,T]

    // ---- carve workspace (aligned to 256B) ----
    const size_t nWp  = (size_t)NL * 4 * ND * ND2;  // packed weight elements
    const size_t nS   = (size_t)NL * NB * ND;       // state elements
    const size_t nAct = (size_t)NL * NB * ND2;      // activation elements per buffer
    const size_t nBi  = (size_t)NL * 4 * ND;
    char*  ws  = (char*)d_ws;
    size_t off = 0;
    auto carve = [&](size_t bytes) -> char* {
        char* p = ws + off;
        off = (off + bytes + 255) & ~(size_t)255;
        return p;
    };
    _Float16* Wp    = (_Float16*)carve(nWp * sizeof(_Float16));
    float*    biasC = (float*)   carve(nBi * sizeof(float));
    float*    cbuf  = (float*)   carve(nS * sizeof(float));
    _Float16* act0  = (_Float16*)carve(nAct * sizeof(_Float16));
    _Float16* act1  = (_Float16*)carve(nAct * sizeof(_Float16));
    _Float16* actb[2] = {act0, act1};
    (void)ws_size; (void)in_sizes; (void)n_in; (void)out_size;

    // ---- one-time (per call) conversions ----
    k_pack_weights<<<8192, 256, 0, stream>>>(W_ih, W_hh, Wp, (int)nWp);
    k_bias_combine<<<(int)((nBi + 255) / 256), 256, 0, stream>>>(b_ih, b_hh, biasC, (int)nBi);
    k_init_act<<<(int)((nS + 255) / 256), 256, 0, stream>>>(hidden, cell, act0, cbuf, (int)nS);

    // ---- 96 autoregressive steps ----
    const dim3 lgrid(NB / 64, ND / 64);
    const size_t strideWp = (size_t)4 * ND * ND2;
    const size_t strideA  = (size_t)NB * ND2;
    const size_t strideC  = (size_t)NB * ND;
    for (int t = 0; t < NTAU; ++t) {
        for (int l = 0; l < NL; ++l) {
            const int lc = (l + 1) % NL;                  // consumer layer of our output x
            const int tc = (l == NL - 1) ? (t + 1) : t;   // consumer step
            k_lstm_layer<<<lgrid, 256, 0, stream>>>(
                actb[t & 1] + (size_t)l * strideA,
                Wp + (size_t)l * strideWp,
                biasC + (size_t)l * 4 * ND,
                cbuf  + (size_t)l * strideC,
                actb[tc & 1] + (size_t)lc * strideA,              // x half of consumer
                actb[(t + 1) & 1] + (size_t)l * strideA + ND);    // own hprev half, next step
        }
        // y_t = last layer's hidden = x-half of layer 0 in the (t+1) buffer
        k_project<<<(NB * NT) / 256, 256, 0, stream>>>(
            actb[(t + 1) & 1], W1, b1, W2, b2, mu, sigma, t);
    }
}
